// SoftImageToReflectanceMap_56221121905131
// MI455X (gfx1250) — compile-verified, matrix operand order checked by reading, not run on hardware
//
#include <hip/hip_runtime.h>
#include <hip/hip_bf16.h>

typedef _Float16 h16 __attribute__((ext_vector_type(16)));
typedef _Float16 h8  __attribute__((ext_vector_type(8)));
typedef __fp16   hp2 __attribute__((ext_vector_type(2)));   // cvt_pkrtz return type
typedef float    f8  __attribute__((ext_vector_type(8)));
typedef float    f2  __attribute__((ext_vector_type(2)));

#define NPIX  16384      // H*W = 128*128
#define NWGB  128        // workgroups per batch
#define LDSBYTES (6 * NPIX * 2)   // Vaug f16, channel-major: 192 KB

__global__ __launch_bounds__(256, 1) void
reflmap_kernel(const float* __restrict__ image,
               const float* __restrict__ nrm,
               const int*   __restrict__ mask,
               const int*   __restrict__ kappa_p,
               float*       __restrict__ out)
{
    extern __shared__ _Float16 sV[];       // [6][NPIX] f16, channel-major
    const int b   = blockIdx.x >> 7;       // batch
    const int wgb = blockIdx.x & (NWGB - 1);
    const int tid = threadIdx.x;

    // ---------- Phase 1: build Vaug = [V*w, V^2*w] (f16) into LDS ----------
    const float* img_b = image + b * 3 * NPIX;
    const int*   msk_b = mask  + b * NPIX;
    for (int i = tid; i < NPIX; i += 256) {
        float w = (msk_b[i] > 0) ? 1.0f : 0.0f;
#pragma unroll
        for (int c = 0; c < 3; ++c) {
            float v = img_b[c * NPIX + i] * w;
            sV[c * NPIX + i]       = (_Float16)v;
            sV[(c + 3) * NPIX + i] = (_Float16)(v * v);
        }
    }
    __syncthreads();

    // ---------- constants: f = coef*exp(-k(1-dp)) = exp2(ea*dp + eb) ----------
    const float kk     = (float)kappa_p[0];
    const float LOG2E  = 1.44269504088896340736f;
    const float coef   = kk / (6.283185307179586f *
                         (1.0f - __builtin_amdgcn_exp2f(-2.0f * kk * LOG2E)));
    const float ea = kk * LOG2E;
    const float eb = -ea + __builtin_amdgcn_logf(coef);   // v_log_f32 is log2

    const int  lane = tid & 31;
    const int  wave = tid >> 5;
    const bool lo   = lane < 16;
    const int  l16  = lane & 15;

    // Per-lane normal-map bases matching the 16x4 f32 A-matrix layout:
    //   lanes 0-15:  V0 = nx (K=0), V1 = ny (K=1)
    //   lanes 16-31: V0 = nz (K=2), V1 = ny (K=3 slot; multiplied by B's zero
    //                K=3 row, so any finite value is fine -> unconditional load,
    //                no divergent branch in the hot loop)
    const float* nx_base = nrm + b * 3 * NPIX + (lo ? 0 : 2 * NPIX);
    const float* ny_base = nrm + b * 3 * NPIX + NPIX;

    for (int r = 0; r < 4; ++r) {
        const int t  = r * (NWGB * 8) + wgb * 8 + wave;   // dir tile 0..4095
        const int d0 = t << 4;

        // B1 (4x16 f32 B-matrix): stereographic directions, computed in-register.
        const int   dir = d0 + l16;
        const float u   = (float)(dir & 255);
        const float v   = (float)(dir >> 8);
        const float p   = 4.0f * ((u + 0.5f) * (1.0f / 256.0f) - 0.5f);
        const float q   = -4.0f * ((v + 0.5f) * (1.0f / 256.0f) - 0.5f);
        const float dd  = 1.0f + p * p + q * q;
        const float inv = __builtin_amdgcn_rcpf(dd);
        const float mx = 2.0f * p * inv, my = 2.0f * q * inv, mz = 2.0f * inv - 1.0f;
        f2 b1;
        b1.x = lo ? mx : mz;     // K=0 / K=2
        b1.y = lo ? my : 0.0f;   // K=1 / K=3 (zero pad row kills A's garbage)

        f8 acc = {};
        for (int kb = 0; kb < 512; ++kb) {
            const int pb = kb * 32;
            h16 b2;                           // f-tile as f16 B-matrix (K=32 pixels)
#pragma unroll
            for (int s = 0; s < 2; ++s) {
                const int pix = pb + s * 16 + l16;
                f2 a1;
                a1.x = nx_base[pix];          // nx (lanes<16) / nz (lanes>=16)
                a1.y = ny_base[pix];          // ny / don't-care (x0 in B)
                f8 zc = {};
                // dp tile: D = (16 pixels) x (16 dirs), K=4 (xyz + pad)
                f8 dp = __builtin_amdgcn_wmma_f32_16x16x4_f32(
                            false, a1, false, b1, (short)0, zc, false, false);
#pragma unroll
                for (int j = 0; j < 4; ++j) {
                    float e0 = __builtin_amdgcn_exp2f(__builtin_fmaf(dp[2*j],     ea, eb));
                    float e1 = __builtin_amdgcn_exp2f(__builtin_fmaf(dp[2*j + 1], ea, eb));
                    hp2 pk = __builtin_amdgcn_cvt_pkrtz(e0, e1);
                    b2[s * 8 + 2*j]     = (_Float16)pk.x;   // D1 layout == B2 layout
                    b2[s * 8 + 2*j + 1] = (_Float16)pk.y;
                }
            }
            // A2 (16x32 f16 A-matrix): Vaug, channel = lane%16 (rows >=6 are
            // garbage but land in discarded D rows). Two ds_load_b128 per lane.
            const int ch    = (l16 < 6) ? l16 : 0;
            const int pbase = pb + (lo ? 0 : 8);
            const h8* lptr  = (const h8*)(sV + ch * NPIX + pbase);
            const h8 alo = lptr[0];   // pixels pbase .. pbase+7   -> K 0-7 / 8-15
            const h8 ahi = lptr[2];   // pixels pbase+16 .. +23    -> K 16-23 / 24-31
            h16 a2;
#pragma unroll
            for (int j = 0; j < 8; ++j) { a2[j] = alo[j]; a2[8 + j] = ahi[j]; }

            acc = __builtin_amdgcn_wmma_f32_16x16x32_f16(
                      false, a2, false, b2, (short)0, acc, false, false);
        }

        // D2: lane = dir (cols 0-15 valid in lanes 0-15), VGPR row = channel.
        if (lo) {
#pragma unroll
            for (int c = 0; c < 3; ++c) {
                float s   = acc[c];
                float num = acc[c + 3];
                out[((b * 3 + c) << 16) + d0 + l16] = num / (s + 0.001f);
            }
        }
    }
}

extern "C" void kernel_launch(void* const* d_in, const int* in_sizes, int n_in,
                              void* d_out, int out_size, void* d_ws, size_t ws_size,
                              hipStream_t stream) {
    (void)in_sizes; (void)n_in; (void)out_size; (void)d_ws; (void)ws_size;
    const float* image = (const float*)d_in[0];
    const float* nrm   = (const float*)d_in[1];
    const int*   mask  = (const int*)d_in[2];
    const int*   kappa = (const int*)d_in[3];
    float* out = (float*)d_out;

    dim3 grid(2 * NWGB);   // 2 batches x 128 workgroups
    dim3 block(256);       // 8 waves
    hipLaunchKernelGGL(reflmap_kernel, grid, block, LDSBYTES, stream,
                       image, nrm, mask, kappa, out);
}
